// YOLOXHead_83245056131535
// MI455X (gfx1250) — matrix operand order repaired and checked by reading, not compiled, not run on hardware
//
#include <hip/hip_runtime.h>
#include <hip/hip_bf16.h>
#include <cstdint>

// ---------------------------------------------------------------------------
// MI455X (gfx1250) YOLOX head. Compute-bound: ~520 GFLOP of 3x3/1x1 convs vs
// ~200MB traffic (9us @ 23.3TB/s). bf16 WMMA implicit GEMM for every conv
// (v_wmma_f32_16x16x32_bf16, f32 accum, fused BN+SiLU), bf16 NHWC tensors.
// K-loop double-buffered: activations via async global->LDS (ASYNCcnt),
// weights via TDM 2D descriptor (TENSORcnt). Block tile 64 pix x 64 cout:
// 2 WMMAs per wave per barrier pair.
// ---------------------------------------------------------------------------

typedef __bf16 bf16_t;
typedef bf16_t v16bf __attribute__((ext_vector_type(16)));
typedef float  v8f   __attribute__((ext_vector_type(8)));
typedef unsigned int u32x4 __attribute__((ext_vector_type(4)));
typedef unsigned int u32x8 __attribute__((ext_vector_type(8)));
typedef int v4i32 __attribute__((__vector_size__(16)));

union FragBF { v16bf v; u32x4 q[2]; };

#if defined(__has_builtin)
#  if __has_builtin(__builtin_amdgcn_global_load_async_to_lds_b128)
#    define HAVE_ASYNC_LDS 1
#  endif
#  if __has_builtin(__builtin_amdgcn_s_wait_asynccnt)
#    define HAVE_WAIT_ASYNC 1
#  endif
#endif
#ifndef HAVE_ASYNC_LDS
#define HAVE_ASYNC_LDS 0
#endif
#ifndef HAVE_WAIT_ASYNC
#define HAVE_WAIT_ASYNC 0
#endif

#define USE_TDM 1

#if HAVE_WAIT_ASYNC
#define WAIT_ASYNCCNT(n) __builtin_amdgcn_s_wait_asynccnt(n)
#else
#define WAIT_ASYNCCNT(n) asm volatile("s_wait_asynccnt " #n :: : "memory")
#endif

__device__ __forceinline__ unsigned short f2bf(float f) {
    unsigned u = __float_as_uint(f);
    unsigned r = ((u >> 16) & 1u) + 0x7FFFu;
    return (unsigned short)((u + r) >> 16);
}
__device__ __forceinline__ float bf2f(unsigned short h) {
    return __uint_as_float(((unsigned)h) << 16);
}
__device__ __forceinline__ float silu(float y) { return y / (1.f + __expf(-y)); }
__device__ __forceinline__ float sigm(float y) { return 1.f / (1.f + __expf(-y)); }

#define NA_TOT 6804
#define PRE 750
#define TOPK 20
#define NC 30

// async global -> LDS, 16 bytes per enabled lane (ASYNCcnt tracked)
__device__ __forceinline__ void async_load_b128(const unsigned short* src,
                                                unsigned short* dst_lds) {
#if HAVE_ASYNC_LDS
    __builtin_amdgcn_global_load_async_to_lds_b128(
        (__attribute__((address_space(1))) v4i32*)src,
        (__attribute__((address_space(3))) v4i32*)dst_lds, 0, 0);
#else
    *(u32x4*)dst_lds = *(const u32x4*)src;
#endif
}

// ---------------------------------------------------------------------------
// TDM 2D tile load: tile_h rows x tile_w_dw dwords, row stride stride_dw,
// contiguous into LDS at lds_off. Issued once per wave (EXEC ignored).
// Descriptor per CDNA5 ISA ch.8 (group0 128b, group1 256b).
// ---------------------------------------------------------------------------
__device__ __forceinline__ void tdm_load_2d(unsigned lds_off, const void* gaddr,
                                            unsigned tile_w_dw, unsigned tile_h,
                                            unsigned stride_dw) {
#if USE_TDM
    unsigned long long ga = (unsigned long long)gaddr;
    u32x4 g0 = {
        1u,                                   // count=1, user mode, no gather
        lds_off,                              // lds_addr [63:32]
        (unsigned)ga,                         // global_addr [95:64]
        (unsigned)((ga >> 32) & 0x1FFFFFFu) | (2u << 30) // addr[56:32] | type=2
    };
    u32x8 g1 = {
        0x20000u,                             // wg_mask=0, data_size=2 (4B)
        (tile_w_dw & 0xFFFFu) << 16,          // tensor_dim0[15:0]
        (tile_w_dw >> 16) | (tile_h << 16),   // tensor_dim0[31:16] | tensor_dim1[15:0]
        (tile_h >> 16) | (tile_w_dw << 16),   // tensor_dim1[31:16] | tile_dim0
        tile_h & 0xFFFFu,                     // tile_dim1 (tile_dim2=0)
        stride_dw,                            // tensor_dim0_stride[31:0]
        0u, 0u
    };
    asm volatile("tensor_load_to_lds %0, %1" :: "s"(g0), "s"(g1) : "memory");
#else
    (void)lds_off; (void)gaddr; (void)tile_w_dw; (void)tile_h; (void)stride_dw;
#endif
}

// ---------------------------------------------------------------------------
// Weight preparation
// ---------------------------------------------------------------------------
__global__ void conv3x3_w_to_bf16(const float* __restrict__ w, // [co][ci][3][3]
                                  unsigned short* __restrict__ wt, // [9][co][ci]
                                  int co, int ci) {
    int idx = blockIdx.x * 256 + threadIdx.x;
    int tot = co * ci * 9;
    if (idx >= tot) return;
    int j = idx % 9;
    int c = (idx / 9) % ci;
    int o = idx / (9 * ci);
    wt[((size_t)j * co + o) * ci + c] = f2bf(w[(size_t)(o * ci + c) * 9 + j]);
}

__global__ void copy_w_to_bf16(const float* __restrict__ w, unsigned short* __restrict__ wt, int n) {
    int idx = blockIdx.x * 256 + threadIdx.x;
    if (idx < n) wt[idx] = f2bf(w[idx]);
}

__global__ void bn_fold(const float* __restrict__ g, const float* __restrict__ b,
                        const float* __restrict__ m, const float* __restrict__ v,
                        float* __restrict__ sc, float* __restrict__ bi) {
    int i = threadIdx.x;
    if (i < 256) {
        float s = g[i] * rsqrtf(v[i] + 1e-3f);
        sc[i] = s;
        bi[i] = b[i] - m[i] * s;
    }
}

// ---------------------------------------------------------------------------
// 3x3 conv, implicit GEMM, bf16 WMMA, fused BN+SiLU. NHWC bf16 in/out.
// Block: 256 thr = 8 waves; tile 64 pixels x 64 cout; wave -> 16x32 (2 WMMA).
// Double-buffered LDS; A via async global->LDS b128, B via TDM (wave 0).
// ---------------------------------------------------------------------------
__global__ __launch_bounds__(256)
void conv3x3_wmma(const unsigned short* __restrict__ in,  // [B][HW][256] bf16
                  const unsigned short* __restrict__ wt,  // [9][256co][256ci] bf16
                  const float* __restrict__ sc, const float* __restrict__ bi,
                  unsigned short* __restrict__ out,       // [B][HW][256] bf16
                  int H, int W) {
    const int HW = H * W;
    const int bidx = blockIdx.z;
    const int pix0 = blockIdx.x * 64;
    const int co0 = blockIdx.y * 64;
    in  += (size_t)bidx * HW * 256;
    out += (size_t)bidx * HW * 256;

    __shared__ __attribute__((aligned(16))) unsigned short As[2][64][32];
    __shared__ __attribute__((aligned(16))) unsigned short Bs[2][64][32];

    const int tid = threadIdx.x;
    const int lane = tid & 31, wid = tid >> 5;
    const int mi = wid & 3, ni = wid >> 2;     // mi: 16-pixel subtile, ni: 32-co half
    const int ap = tid >> 2, aseg = tid & 3;
    const int apix = pix0 + ap;
    const int ay = apix / W, ax = apix % W;

    v8f acc[2];
    acc[0] = (v8f){0.f, 0.f, 0.f, 0.f, 0.f, 0.f, 0.f, 0.f};
    acc[1] = (v8f){0.f, 0.f, 0.f, 0.f, 0.f, 0.f, 0.f, 0.f};

    // stage A tile (64 pixels x 32 ci) for chunk c into buffer buf
    auto stageA = [&](int c, int buf) {
        const int j = c >> 3, cb = c & 7;
        const int ky = j / 3 - 1, kx = j % 3 - 1;
        const int iy = ay + ky, ix = ax + kx;
        const bool inb = (apix < HW) && (iy >= 0) && (iy < H) && (ix >= 0) && (ix < W);
        unsigned short* dst = &As[buf][ap][aseg * 8];
        if (inb) {
            async_load_b128(in + (size_t)(iy * W + ix) * 256 + cb * 32 + aseg * 8, dst);
        } else {
            u32x4 z = {0u, 0u, 0u, 0u};
            *(u32x4*)dst = z;
        }
    };
    // stage B tile (64 co x 32 ci = 64 rows x 64B, row stride 512B) via TDM
    auto stageB = [&](int c, int buf) {
#if USE_TDM
        const int j = c >> 3, cb = c & 7;
        const unsigned short* gsrc = wt + (((size_t)j * 256 + co0) * 256) + cb * 32;
        unsigned lds_off =
            (unsigned)(size_t)(__attribute__((address_space(3))) char*)&Bs[buf][0][0];
        tdm_load_2d(lds_off, gsrc, /*tile_w_dw=*/16, /*tile_h=*/64, /*stride_dw=*/128);
#else
        const int j = c >> 3, cb = c & 7;
        int r = tid >> 2, seg = tid & 3;
        *(u32x4*)&Bs[buf][r][seg * 8] =
            *(const u32x4*)(wt + (((size_t)j * 256 + (co0 + r)) * 256) + cb * 32 + seg * 8);
#endif
    };

    stageA(0, 0);
    if (wid == 0) stageB(0, 0);

    for (int c = 0; c < 72; ++c) {
        const int buf = c & 1;
        if (c + 1 < 72) {
            stageA(c + 1, buf ^ 1);
            if (wid == 0) stageB(c + 1, buf ^ 1);
#if HAVE_ASYNC_LDS
            WAIT_ASYNCCNT(1);   // chunk c's async A-loads complete
#endif
            if (wid == 0) __builtin_amdgcn_s_wait_tensorcnt(1); // chunk c's TDM done
        } else {
#if HAVE_ASYNC_LDS
            WAIT_ASYNCCNT(0);
#endif
            if (wid == 0) __builtin_amdgcn_s_wait_tensorcnt(0);
        }
        __syncthreads();

        FragBF fa, fb0, fb1;
        {
            int row = mi * 16 + (lane & 15);
            int kb = (lane & 16) ? 8 : 0;
            fa.q[0] = *(const u32x4*)&As[buf][row][kb];
            fa.q[1] = *(const u32x4*)&As[buf][row][kb + 16];
        }
        {
            int kb = (lane & 16) ? 8 : 0;
            int row0 = ni * 32 + (lane & 15);
            fb0.q[0] = *(const u32x4*)&Bs[buf][row0][kb];
            fb0.q[1] = *(const u32x4*)&Bs[buf][row0][kb + 16];
            int row1 = row0 + 16;
            fb1.q[0] = *(const u32x4*)&Bs[buf][row1][kb];
            fb1.q[1] = *(const u32x4*)&Bs[buf][row1][kb + 16];
        }
        acc[0] = __builtin_amdgcn_wmma_f32_16x16x32_bf16(
            false, fa.v, false, fb0.v, (short)0, acc[0], false, false);
        acc[1] = __builtin_amdgcn_wmma_f32_16x16x32_bf16(
            false, fa.v, false, fb1.v, (short)0, acc[1], false, false);
        __syncthreads();   // all waves done reading buf before it is restaged
    }

    const int n = lane & 15;
    const int mo = (lane & 16) ? 8 : 0;
    #pragma unroll
    for (int t = 0; t < 2; ++t) {
        const int co = co0 + ni * 32 + t * 16 + n;
        const float s = sc[co], bb = bi[co];
        #pragma unroll
        for (int e = 0; e < 8; ++e) {
            int pix = pix0 + mi * 16 + e + mo;
            if (pix < HW) {
                float y = acc[t][e] * s + bb;
                out[(size_t)pix * 256 + co] = f2bf(silu(y));
            }
        }
    }
}

// ---------------------------------------------------------------------------
// Stem 1x1 conv: fp32 NCHW input (cin variable) -> bf16 NHWC, BN+SiLU, WMMA.
// ---------------------------------------------------------------------------
__global__ __launch_bounds__(256)
void stem_conv1x1_wmma(const float* __restrict__ in,          // [B][cin][HW] f32
                       const unsigned short* __restrict__ wt, // [256co][cin] bf16
                       const float* __restrict__ sc, const float* __restrict__ bi,
                       unsigned short* __restrict__ out,      // [B][HW][256] bf16
                       int HW, int cin) {
    const int bidx = blockIdx.z;
    const int pix0 = blockIdx.x * 64;
    const int co0 = blockIdx.y * 32;
    in  += (size_t)bidx * cin * HW;
    out += (size_t)bidx * HW * 256;

    __shared__ __attribute__((aligned(16))) unsigned short As[64][32];
    __shared__ __attribute__((aligned(16))) unsigned short Bs[32][32];

    const int tid = threadIdx.x;
    const int lane = tid & 31, wid = tid >> 5;
    const int mi = wid & 3, ni = wid >> 2;

    v8f acc = {0.f, 0.f, 0.f, 0.f, 0.f, 0.f, 0.f, 0.f};

    const int ap = tid >> 2, aseg = tid & 3;
    const int apix = pix0 + ap;
    const bool inb = apix < HW;
    const int nkc = cin / 32;

    for (int cb = 0; cb < nkc; ++cb) {
        {
            int ci0 = cb * 32 + aseg * 8;
            unsigned short tmp[8];
            #pragma unroll
            for (int i = 0; i < 8; ++i)
                tmp[i] = inb ? f2bf(in[(size_t)(ci0 + i) * HW + apix]) : (unsigned short)0;
            #pragma unroll
            for (int i = 0; i < 8; ++i) As[ap][aseg * 8 + i] = tmp[i];
        }
        if (tid < 128) {
            int r = tid >> 2, seg = tid & 3;
            const unsigned short* src = wt + (size_t)(co0 + r) * cin + cb * 32 + seg * 8;
            __builtin_prefetch(src + 32, 0, 0); // -> global_prefetch_b8
            *(u32x4*)&Bs[r][seg * 8] = *(const u32x4*)src;
        }
        __syncthreads();

        FragBF fa, fb;
        {
            int row = mi * 16 + (lane & 15);
            int kb = (lane & 16) ? 8 : 0;
            fa.q[0] = *(const u32x4*)&As[row][kb];
            fa.q[1] = *(const u32x4*)&As[row][kb + 16];
        }
        {
            int row = ni * 16 + (lane & 15);
            int kb = (lane & 16) ? 8 : 0;
            fb.q[0] = *(const u32x4*)&Bs[row][kb];
            fb.q[1] = *(const u32x4*)&Bs[row][kb + 16];
        }
        acc = __builtin_amdgcn_wmma_f32_16x16x32_bf16(
            false, fa.v, false, fb.v, (short)0, acc, false, false);
        __syncthreads();
    }

    const int n = lane & 15;
    const int co = co0 + ni * 16 + n;
    const float s = sc[co], bb = bi[co];
    const int mo = (lane & 16) ? 8 : 0;
    for (int e = 0; e < 8; ++e) {
        int pix = pix0 + mi * 16 + e + mo;
        if (pix < HW) {
            float y = acc[e] * s + bb;
            out[(size_t)pix * 256 + co] = f2bf(silu(y));
        }
    }
}

// ---------------------------------------------------------------------------
// 1x1 prediction heads + decode into dec[B][NA_TOT][35]
// ---------------------------------------------------------------------------
__global__ void pred_decode(const unsigned short* __restrict__ rf,
                            const unsigned short* __restrict__ cf,
                            const float* __restrict__ wreg, const float* __restrict__ breg,
                            const float* __restrict__ wobj, const float* __restrict__ bobj,
                            const float* __restrict__ wcls, const float* __restrict__ bcls,
                            float* __restrict__ dec,
                            int H, int W, int aoff, float stridef) {
    const int HW = H * W;
    int idx = blockIdx.x * 256 + threadIdx.x;
    if (idx >= 8 * HW) return;
    int b = idx / HW, pix = idx % HW;
    const unsigned short* rrow = rf + ((size_t)b * HW + pix) * 256;
    const unsigned short* crow = cf + ((size_t)b * HW + pix) * 256;

    float a0 = breg[0], a1 = breg[1], a2 = breg[2], a3 = breg[3], ao = bobj[0];
    for (int c = 0; c < 256; ++c) {
        float v = bf2f(rrow[c]);
        a0 += v * wreg[c];
        a1 += v * wreg[256 + c];
        a2 += v * wreg[512 + c];
        a3 += v * wreg[768 + c];
        ao += v * wobj[c];
    }
    float ac[NC];
    for (int o = 0; o < NC; ++o) ac[o] = bcls[o];
    for (int c = 0; c < 256; ++c) {
        float v = bf2f(crow[c]);
        for (int o = 0; o < NC; ++o) ac[o] += v * wcls[o * 256 + c];
    }
    float* drow = dec + ((size_t)b * NA_TOT + aoff + pix) * 35;
    int gy = pix / W, gx = pix % W;
    drow[0] = (a0 + (float)gx) * stridef;
    drow[1] = (a1 + (float)gy) * stridef;
    drow[2] = __expf(a2) * stridef;
    drow[3] = __expf(a3) * stridef;
    drow[4] = sigm(ao);
    for (int o = 0; o < NC; ++o) drow[5 + o] = sigm(ac[o]);
}

// ---------------------------------------------------------------------------
// Post-processing: top-750 by obj, sort by score, NMS, top-20. 1 block/image.
// ---------------------------------------------------------------------------
__device__ __forceinline__ float iou_box(const float* a, const float* b) {
    float tlx = fmaxf(a[0], b[0]), tly = fmaxf(a[1], b[1]);
    float brx = fminf(a[2], b[2]), bry = fminf(a[3], b[3]);
    float w = fmaxf(brx - tlx, 0.f), h = fmaxf(bry - tly, 0.f);
    float inter = w * h;
    float aa = (a[2] - a[0]) * (a[3] - a[1]);
    float ab = (b[2] - b[0]) * (b[3] - b[1]);
    return inter / (aa + ab - inter + 1e-9f);
}

__global__ __launch_bounds__(256)
void postpro(const float* __restrict__ dec, float* __restrict__ out_pred,
             int* __restrict__ gidx, float* __restrict__ cls_sc, float* __restrict__ fg_sc) {
    const int b = blockIdx.x;
    const int tid = threadIdx.x;
    const float* D = dec + (size_t)b * NA_TOT * 35;

    __shared__ unsigned key[8192];
    __shared__ int aidx[768];
    __shared__ float bx[PRE][4];
    __shared__ float conf[768];
    __shared__ int klass[768];
    __shared__ float ssc[768];
    __shared__ unsigned skeep[32];
    __shared__ int sdead;
    __shared__ float rmax[256];
    __shared__ int rarg[256];

    // stage 1: pack obj|idx, ascending bitonic sort over 8192
    for (int a = tid; a < 8192; a += 256) {
        unsigned kk = 0;
        if (a < NA_TOT) kk = (__float_as_uint(D[a * 35 + 4]) & 0xFFFFE000u) | (unsigned)a;
        key[a] = kk;
    }
    __syncthreads();
    for (unsigned k2 = 2; k2 <= 8192; k2 <<= 1) {
        for (unsigned jj = k2 >> 1; jj > 0; jj >>= 1) {
            for (int i = tid; i < 8192; i += 256) {
                int ixj = i ^ (int)jj;
                if (ixj > i) {
                    bool up = ((i & k2) == 0);
                    unsigned a = key[i], c = key[ixj];
                    if ((a > c) == up) { key[i] = c; key[ixj] = a; }
                }
            }
            __syncthreads();
        }
    }

    // stage 2: scores for top-750
    for (int r = tid; r < PRE; r += 256) {
        int a = (int)(key[8191 - r] & 8191u);
        if (a >= NA_TOT) a = 0;
        float obj = D[a * 35 + 4];
        float bp = -1e30f; int bc = 0;
        for (int c = 0; c < NC; ++c) {
            float p = D[a * 35 + 5 + c];
            if (p > bp) { bp = p; bc = c; }
        }
        aidx[r] = a; conf[r] = bp; klass[r] = bc; ssc[r] = obj * bp;
    }
    __syncthreads();
    for (int r = tid; r < 1024; r += 256) {
        unsigned kk = 0;
        if (r < PRE) kk = (__float_as_uint(ssc[r]) & 0xFFFFFC00u) | (unsigned)r;
        key[r] = kk;
    }
    __syncthreads();
    for (unsigned k2 = 2; k2 <= 1024; k2 <<= 1) {
        for (unsigned jj = k2 >> 1; jj > 0; jj >>= 1) {
            for (int i = tid; i < 1024; i += 256) {
                int ixj = i ^ (int)jj;
                if (ixj > i) {
                    bool up = ((i & k2) == 0);
                    unsigned a = key[i], c = key[ixj];
                    if ((a > c) == up) { key[i] = c; key[ixj] = a; }
                }
            }
            __syncthreads();
        }
    }
    // permute arrays into score order
    float pssc[3], pconf[3], pbox[3][4];
    int paidx[3], pkl[3];
    int cnt = 0;
    for (int q = tid; q < PRE; q += 256) {
        int r = (int)(key[1023 - q] & 1023u);
        if (r >= PRE) r = 0;
        pssc[cnt] = ssc[r]; pconf[cnt] = conf[r]; paidx[cnt] = aidx[r]; pkl[cnt] = klass[r];
        int a = aidx[r];
        float cx = D[a * 35 + 0], cy = D[a * 35 + 1];
        float hw = D[a * 35 + 2] * 0.5f, hh = D[a * 35 + 3] * 0.5f;
        pbox[cnt][0] = cx - hw; pbox[cnt][1] = cy - hh;
        pbox[cnt][2] = cx + hw; pbox[cnt][3] = cy + hh;
        cnt++;
    }
    __syncthreads();
    cnt = 0;
    for (int q = tid; q < PRE; q += 256) {
        ssc[q] = pssc[cnt]; conf[q] = pconf[cnt]; aidx[q] = paidx[cnt]; klass[q] = pkl[cnt];
        bx[q][0] = pbox[cnt][0]; bx[q][1] = pbox[cnt][1];
        bx[q][2] = pbox[cnt][2]; bx[q][3] = pbox[cnt][3];
        cnt++;
    }
    if (tid < 32) skeep[tid] = 0xFFFFFFFFu;
    __syncthreads();

    // NMS (sequential over i, parallel over earlier j)
    for (int i = 1; i < PRE; ++i) {
        if (tid == 0) sdead = 0;
        __syncthreads();
        int dead = 0;
        for (int j = tid; j < i; j += 256) {
            if (((skeep[j >> 5] >> (j & 31)) & 1u) && klass[j] == klass[i] &&
                iou_box(bx[j], bx[i]) > 0.75f) dead = 1;
        }
        if (dead) atomicOr(&sdead, 1);
        __syncthreads();
        if (tid == 0) {
            unsigned m = 1u << (i & 31);
            if (sdead) skeep[i >> 5] &= ~m; else skeep[i >> 5] |= m;
        }
        __syncthreads();
    }

    // top-20 of effective scores
    float* eff = reinterpret_cast<float*>(key);
    for (int q = tid; q < PRE; q += 256) {
        bool kp = ((skeep[q >> 5] >> (q & 31)) & 1u) != 0;
        eff[q] = kp ? ssc[q] : ssc[q] - 1e9f;
    }
    __syncthreads();
    for (int t = 0; t < TOPK; ++t) {
        float lm = -3e9f; int la = 0;
        for (int q = tid; q < PRE; q += 256)
            if (eff[q] > lm) { lm = eff[q]; la = q; }
        rmax[tid] = lm; rarg[tid] = la;
        __syncthreads();
        for (int s = 128; s > 0; s >>= 1) {
            if (tid < s && rmax[tid + s] > rmax[tid]) {
                rmax[tid] = rmax[tid + s]; rarg[tid] = rarg[tid + s];
            }
            __syncthreads();
        }
        if (tid == 0) {
            int q = rarg[0];
            int g = aidx[q];
            float* o = out_pred + ((size_t)(b * TOPK + t)) * 37;
            o[0] = bx[q][0]; o[1] = bx[q][1]; o[2] = bx[q][2]; o[3] = bx[q][3];
            float obj = D[g * 35 + 4];
            o[4] = obj;
            o[5] = conf[q];
            o[6] = (float)klass[q];
            for (int c = 0; c < NC; ++c) o[7 + c] = D[g * 35 + 5 + c];
            gidx[b * TOPK + t] = g;
            cls_sc[b * TOPK + t] = conf[q];
            fg_sc[b * TOPK + t] = obj;
            eff[q] = -2e9f;
        }
        __syncthreads();
    }
}

// ---------------------------------------------------------------------------
// Gather per-token features (bf16 NHWC -> f32)
// ---------------------------------------------------------------------------
__global__ void gather_feats(const unsigned short* cf0, const unsigned short* cf1,
                             const unsigned short* cf2, const unsigned short* rf0,
                             const unsigned short* rf1, const unsigned short* rf2,
                             const int* __restrict__ gidx,
                             float* __restrict__ fcls, float* __restrict__ freg) {
    int t = blockIdx.x;  // 160
    int c = threadIdx.x; // 256
    int b = t / TOPK;
    int a = gidx[t];
    const unsigned short *pc, *pr;
    if (a < 5184) {
        pc = cf0 + ((size_t)b * 5184 + a) * 256;
        pr = rf0 + ((size_t)b * 5184 + a) * 256;
    } else if (a < 6480) {
        pc = cf1 + ((size_t)b * 1296 + (a - 5184)) * 256;
        pr = rf1 + ((size_t)b * 1296 + (a - 5184)) * 256;
    } else {
        pc = cf2 + ((size_t)b * 324 + (a - 6480)) * 256;
        pr = rf2 + ((size_t)b * 324 + (a - 6480)) * 256;
    }
    fcls[(size_t)t * 256 + c] = bf2f(pc[c]);
    freg[(size_t)t * 256 + c] = bf2f(pr[c]);
}

// ---------------------------------------------------------------------------
// Small fp32 GEMMs / MSA helpers
// ---------------------------------------------------------------------------
__global__ void gemm_xwT(const float* __restrict__ x, const float* __restrict__ w,
                         const float* __restrict__ bias, float* __restrict__ out,
                         int M, int N, int K, int ldo) {
    int idx = blockIdx.x * 256 + threadIdx.x;
    if (idx >= M * N) return;
    int m = idx / N, n = idx % N;
    float acc = bias ? bias[n] : 0.f;
    const float* xr = x + (size_t)m * K;
    const float* wr = w + (size_t)n * K;
    for (int k = 0; k < K; ++k) acc += xr[k] * wr[k];
    out[(size_t)m * ldo + n] = acc;
}

__global__ void gemm_nn(const float* __restrict__ a, const float* __restrict__ bm,
                        float* __restrict__ out, int M, int N, int K, int ldb, int ldo) {
    int idx = blockIdx.x * 256 + threadIdx.x;
    if (idx >= M * N) return;
    int m = idx / N, n = idx % N;
    float acc = 0.f;
    const float* ar = a + (size_t)m * K;
    for (int k = 0; k < K; ++k) acc += ar[k] * bm[(size_t)k * ldb + n];
    out[(size_t)m * ldo + n] = acc;
}

__global__ void copy_strided(const float* __restrict__ src, float* __restrict__ dst,
                             int rows, int cols, int lds, int ldd) {
    int idx = blockIdx.x * 256 + threadIdx.x;
    if (idx >= rows * cols) return;
    int r = idx / cols, c = idx % cols;
    dst[(size_t)r * ldd + c] = src[(size_t)r * lds + c];
}

// logits[h][n][m] = 25 * score[m] * cos(q_n, k_m)
__global__ void attn_scores(const float* __restrict__ y, const float* __restrict__ score,
                            float* __restrict__ logits) {
    int idx = blockIdx.x * 256 + threadIdx.x;
    if (idx >= 4 * 160 * 160) return;
    int m = idx % 160, n = (idx / 160) % 160, h = idx / 25600;
    const float* q = y + (size_t)n * 768 + h * 64;
    const float* k = y + (size_t)m * 768 + 256 + h * 64;
    float d = 0.f, qq = 0.f, kk = 0.f;
    for (int i = 0; i < 64; ++i) { d += q[i] * k[i]; qq += q[i] * q[i]; kk += k[i] * k[i]; }
    logits[idx] = 25.f * score[m] * d * rsqrtf(qq * kk + 1e-30f);
}

__global__ void softmax_rows(float* __restrict__ mat, int rows, int cols) {
    int r = blockIdx.x * 64 + threadIdx.x;
    if (r >= rows) return;
    float* row = mat + (size_t)r * cols;
    float mx = -3e38f;
    for (int c = 0; c < cols; ++c) mx = fmaxf(mx, row[c]);
    float s = 0.f;
    for (int c = 0; c < cols; ++c) { float e = __expf(row[c] - mx); row[c] = e; s += e; }
    float inv = 1.f / s;
    for (int c = 0; c < cols; ++c) row[c] *= inv;
}

__global__ void avg2(const float* a, const float* b, float* o, int n) {
    int i = blockIdx.x * 256 + threadIdx.x;
    if (i < n) o[i] = 0.5f * (a[i] + b[i]);
}

// tmix[n][0:256]=attn@v, tmix[n][256:512]=v(n)
__global__ void mix_kernel(const float* __restrict__ attn, const float* __restrict__ y,
                           float* __restrict__ tmix) {
    int idx = blockIdx.x * 256 + threadIdx.x;
    if (idx >= 160 * 256) return;
    int n = idx / 256, o = idx % 256;
    int h = o / 64;
    const float* ar = attn + (size_t)h * 25600 + n * 160;
    float acc = 0.f;
    for (int m = 0; m < 160; ++m) acc += ar[m] * y[(size_t)m * 768 + 512 + o];
    tmix[(size_t)n * 512 + o] = acc;
    tmix[(size_t)n * 512 + 256 + o] = y[(size_t)n * 768 + 512 + o];
}

__global__ void sim_raw(const float* __restrict__ ycls, float* __restrict__ simr) {
    int idx = blockIdx.x * 256 + threadIdx.x;
    if (idx >= 160 * 160) return;
    int n = idx / 160, m = idx % 160;
    float acc = 0.f;
    for (int h = 0; h < 4; ++h) {
        const float* vn = ycls + (size_t)n * 768 + 512 + h * 64;
        const float* vm = ycls + (size_t)m * 768 + 512 + h * 64;
        float d = 0.f, nn = 0.f, mm = 0.f;
        for (int i = 0; i < 64; ++i) { d += vn[i] * vm[i]; nn += vn[i] * vn[i]; mm += vm[i] * vm[i]; }
        acc += d * rsqrtf(nn * mm + 1e-30f);
    }
    simr[idx] = acc * 0.25f;
}

__global__ void sim_r2(const float* __restrict__ attn, const float* __restrict__ simr,
                       float* __restrict__ s2) {
    int n = blockIdx.x * 32 + threadIdx.x;
    if (n >= 160) return;
    float row[160];
    float mx = -3e38f;
    for (int m = 0; m < 160; ++m) {
        float ab = 0.25f * (attn[n * 160 + m] + attn[25600 + n * 160 + m] +
                            attn[51200 + n * 160 + m] + attn[76800 + n * 160 + m]);
        row[m] = ab;
        mx = fmaxf(mx, ab);
    }
    float s = 0.f;
    for (int m = 0; m < 160; ++m) { row[m] = __expf(row[m] - mx); s += row[m]; }
    float inv = 1.f / s, tot = 0.f;
    for (int m = 0; m < 160; ++m) {
        float v = row[m] * inv * (simr[n * 160 + m] > 0.75f ? 1.f : 0.f);
        row[m] = v; tot += v;
    }
    float inv2 = 1.f / (tot + 1e-30f);
    for (int m = 0; m < 160; ++m) s2[(size_t)n * 160 + m] = row[m] * inv2;
}

// fc (8,20,30) and rfc (8,20,1) into d_out tail
__global__ void final_heads(const float* __restrict__ o2c, const float* __restrict__ o2r,
                            const float* __restrict__ wlp, const float* __restrict__ blp,
                            const float* __restrict__ wrlp, const float* __restrict__ brlp,
                            float* __restrict__ out) {
    int idx = blockIdx.x * 256 + threadIdx.x;
    if (idx >= 160 * 32) return;
    int t = idx / 32, o = idx % 32;
    if (o < 31) {
        float acc = blp[o];
        const float* xr = o2c + (size_t)t * 1024;
        const float* wr = wlp + (size_t)o * 1024;
        for (int k = 0; k < 1024; ++k) acc += xr[k] * wr[k];
        if (o < NC) out[8 * TOPK * 37 + t * NC + o] = acc;
    } else {
        float acc = brlp[0];
        const float* xr = o2r + (size_t)t * 1024;
        for (int k = 0; k < 1024; ++k) acc += xr[k] * wrlp[k];
        out[8 * TOPK * 37 + 8 * TOPK * NC + t] = acc;
    }
}

// ---------------------------------------------------------------------------
// Host orchestration
// ---------------------------------------------------------------------------
extern "C" void kernel_launch(void* const* d_in, const int* in_sizes, int n_in,
                              void* d_out, int out_size, void* d_ws, size_t ws_size,
                              hipStream_t stream) {
    (void)in_sizes; (void)n_in; (void)out_size; (void)ws_size;

    const int Hs[3] = {72, 36, 18}, Ws[3] = {72, 36, 18};
    const int HWs[3] = {5184, 1296, 324};
    const int CIN[3] = {256, 512, 1024};
    const float STR[3] = {8.f, 16.f, 32.f};
    const int AOFF[3] = {0, 5184, 6480};
    // conv_bn leaf order: b,g,m,v,w ; conv order: reg1a,reg1b,reg2a,reg2b,cls1a,cls1b,cls2a,cls2b
    const int CONV_OFF[8] = {24, 29, 34, 39, 0, 5, 10, 15};

    auto LIN = [&](int lvl, int off) { return (const float*)d_in[4 + lvl * 51 + off]; };

    size_t cur = 0;
    auto take = [&](size_t bytes) {
        size_t o = cur;
        cur += (bytes + 255) & ~(size_t)255;
        return o;
    };
    auto WSU = [&](size_t off) { return (unsigned short*)((char*)d_ws + off); };
    auto WSF = [&](size_t off) { return (float*)((char*)d_ws + off); };
    auto WSI = [&](size_t off) { return (int*)((char*)d_ws + off); };

    size_t oW3 = take(24ull * 9 * 256 * 256 * 2);
    size_t oW1[3];
    for (int l = 0; l < 3; ++l) oW1[l] = take((size_t)CIN[l] * 256 * 2);
    size_t oSC = take(27ull * 256 * 4);
    size_t oBI = take(27ull * 256 * 4);
    const size_t FEAT0 = 8ull * 5184 * 256 * 2;
    size_t oX = take(FEAT0), oT1 = take(FEAT0), oRF = take(FEAT0), oCF = take(FEAT0);
    size_t oRF2[3], oCF2[3];
    for (int l = 0; l < 3; ++l) oRF2[l] = take(8ull * HWs[l] * 256 * 2);
    for (int l = 0; l < 3; ++l) oCF2[l] = take(8ull * HWs[l] * 256 * 2);
    size_t oDEC = take(8ull * NA_TOT * 35 * 4);
    size_t oGIDX = take(160 * 4), oCSC = take(160 * 4), oFGS = take(160 * 4);
    size_t oFC = take(160ull * 256 * 4), oFR = take(160ull * 256 * 4);
    size_t oYC = take(160ull * 768 * 4), oYR = take(160ull * 768 * 4);
    size_t oLC = take(4ull * 160 * 160 * 4), oLR = take(4ull * 160 * 160 * 4);
    size_t oAT = take(4ull * 160 * 160 * 4);
    size_t oTMC = take(160ull * 512 * 4), oTMR = take(160ull * 512 * 4);
    size_t oSIM = take(160ull * 160 * 4), oS2 = take(160ull * 160 * 4);
    size_t oFBC = take(160ull * 512 * 4), oFBR = take(160ull * 512 * 4);
    size_t oGC = take(160ull * 1024 * 4), oGR = take(160ull * 1024 * 4);
    size_t oO2C = take(160ull * 1024 * 4), oO2R = take(160ull * 1024 * 4);

    // ---- weight prep ----
    for (int l = 0; l < 3; ++l) {
        for (int k = 0; k < 8; ++k) {
            int base = CONV_OFF[k];
            int id = l * 8 + k;
            conv3x3_w_to_bf16<<<(256 * 256 * 9 + 255) / 256, 256, 0, stream>>>(
                LIN(l, base + 4), WSU(oW3) + (size_t)id * 9 * 256 * 256, 256, 256);
            bn_fold<<<1, 256, 0, stream>>>(LIN(l, base + 1), LIN(l, base + 0),
                                           LIN(l, base + 2), LIN(l, base + 3),
                                           WSF(oSC) + id * 256, WSF(oBI) + id * 256);
        }
        int n = 256 * CIN[l];
        copy_w_to_bf16<<<(n + 255) / 256, 256, 0, stream>>>(LIN(l, 46 + 4), WSU(oW1[l]), n);
        int sid = 24 + l;
        bn_fold<<<1, 256, 0, stream>>>(LIN(l, 46 + 1), LIN(l, 46 + 0), LIN(l, 46 + 2),
                                       LIN(l, 46 + 3), WSF(oSC) + sid * 256, WSF(oBI) + sid * 256);
    }

    // ---- per-level conv stacks ----
    for (int l = 0; l < 3; ++l) {
        const int HW = HWs[l];
        dim3 sgrid((HW + 63) / 64, 8, 8);   // stem: 32-co tiles
        dim3 cgrid((HW + 63) / 64, 4, 8);   // conv3x3: 64-co tiles
        int sid = 24 + l;
        stem_conv1x1_wmma<<<sgrid, 256, 0, stream>>>(
            (const float*)d_in[l], WSU(oW1[l]), WSF(oSC) + sid * 256, WSF(oBI) + sid * 256,
            WSU(oX), HW, CIN[l]);

        auto C3 = [&](int k, size_t src, size_t dst) {
            int id = l * 8 + k;
            conv3x3_wmma<<<cgrid, 256, 0, stream>>>(
                WSU(src), WSU(oW3) + (size_t)id * 9 * 256 * 256,
                WSF(oSC) + id * 256, WSF(oBI) + id * 256, WSU(dst), Hs[l], Ws[l]);
        };
        C3(0, oX, oT1);  C3(1, oT1, oRF);       // rf
        C3(2, oX, oT1);  C3(3, oT1, oRF2[l]);   // rf2
        C3(4, oX, oT1);  C3(5, oT1, oCF);       // cf
        C3(6, oX, oT1);  C3(7, oT1, oCF2[l]);   // cf2

        pred_decode<<<(8 * HW + 255) / 256, 256, 0, stream>>>(
            WSU(oRF), WSU(oCF),
            LIN(l, 45), LIN(l, 44),   // reg_pred w,b
            LIN(l, 23), LIN(l, 22),   // obj_pred w,b
            LIN(l, 21), LIN(l, 20),   // cls_pred w,b
            WSF(oDEC), Hs[l], Ws[l], AOFF[l], STR[l]);
    }

    // ---- postpro ----
    postpro<<<8, 256, 0, stream>>>(WSF(oDEC), (float*)d_out, WSI(oGIDX), WSF(oCSC), WSF(oFGS));

    gather_feats<<<160, 256, 0, stream>>>(WSU(oCF2[0]), WSU(oCF2[1]), WSU(oCF2[2]),
                                          WSU(oRF2[0]), WSU(oRF2[1]), WSU(oRF2[2]),
                                          WSI(oGIDX), WSF(oFC), WSF(oFR));

    // ---- MSA ----
    const float* qkv_cls_w = (const float*)d_in[169];
    const float* qkv_reg_w = (const float*)d_in[170];
    gemm_xwT<<<(160 * 768 + 255) / 256, 256, 0, stream>>>(WSF(oFC), qkv_cls_w, nullptr,
                                                          WSF(oYC), 160, 768, 256, 768);
    gemm_xwT<<<(160 * 768 + 255) / 256, 256, 0, stream>>>(WSF(oFR), qkv_reg_w, nullptr,
                                                          WSF(oYR), 160, 768, 256, 768);
    attn_scores<<<(102400 + 255) / 256, 256, 0, stream>>>(WSF(oYC), WSF(oCSC), WSF(oLC));
    attn_scores<<<(102400 + 255) / 256, 256, 0, stream>>>(WSF(oYR), WSF(oFGS), WSF(oLR));
    softmax_rows<<<(640 + 63) / 64, 64, 0, stream>>>(WSF(oLC), 640, 160);
    softmax_rows<<<(640 + 63) / 64, 64, 0, stream>>>(WSF(oLR), 640, 160);
    avg2<<<(102400 + 255) / 256, 256, 0, stream>>>(WSF(oLC), WSF(oLR), WSF(oAT), 102400);
    mix_kernel<<<(160 * 256 + 255) / 256, 256, 0, stream>>>(WSF(oAT), WSF(oYC), WSF(oTMC));
    mix_kernel<<<(160 * 256 + 255) / 256, 256, 0, stream>>>(WSF(oAT), WSF(oYR), WSF(oTMR));
    sim_raw<<<(25600 + 255) / 256, 256, 0, stream>>>(WSF(oYC), WSF(oSIM));
    sim_r2<<<5, 32, 0, stream>>>(WSF(oAT), WSF(oSIM), WSF(oS2));

    const float* l1c_w = (const float*)d_in[162]; const float* l1c_b = (const float*)d_in[161];
    const float* l1r_w = (const float*)d_in[164]; const float* l1r_b = (const float*)d_in[163];
    const float* l2c_w = (const float*)d_in[166]; const float* l2c_b = (const float*)d_in[165];
    const float* l2r_w = (const float*)d_in[168]; const float* l2r_b = (const float*)d_in[167];
    gemm_xwT<<<(160 * 512 + 255) / 256, 256, 0, stream>>>(WSF(oTMC), l1c_w, l1c_b,
                                                          WSF(oFBC), 160, 512, 512, 512);
    gemm_xwT<<<(160 * 512 + 255) / 256, 256, 0, stream>>>(WSF(oTMR), l1r_w, l1r_b,
                                                          WSF(oFBR), 160, 512, 512, 512);
    gemm_nn<<<(160 * 512 + 255) / 256, 256, 0, stream>>>(WSF(oS2), WSF(oFBC), WSF(oGC),
                                                         160, 512, 160, 512, 1024);
    gemm_nn<<<(160 * 512 + 255) / 256, 256, 0, stream>>>(WSF(oS2), WSF(oFBR), WSF(oGR),
                                                         160, 512, 160, 512, 1024);
    copy_strided<<<(160 * 512 + 255) / 256, 256, 0, stream>>>(WSF(oFBC), WSF(oGC) + 512,
                                                              160, 512, 512, 1024);
    copy_strided<<<(160 * 512 + 255) / 256, 256, 0, stream>>>(WSF(oFBR), WSF(oGR) + 512,
                                                              160, 512, 512, 1024);
    gemm_xwT<<<(160 * 1024 + 255) / 256, 256, 0, stream>>>(WSF(oGC), l2c_w, l2c_b,
                                                           WSF(oO2C), 160, 1024, 1024, 1024);
    gemm_xwT<<<(160 * 1024 + 255) / 256, 256, 0, stream>>>(WSF(oGR), l2r_w, l2r_b,
                                                           WSF(oO2R), 160, 1024, 1024, 1024);

    const float* lp_w = (const float*)d_in[158];  const float* lp_b = (const float*)d_in[157];
    const float* rlp_w = (const float*)d_in[160]; const float* rlp_b = (const float*)d_in[159];
    final_heads<<<(160 * 32 + 255) / 256, 256, 0, stream>>>(WSF(oO2C), WSF(oO2R),
                                                            lp_w, lp_b, rlp_w, rlp_b,
                                                            (float*)d_out);
}
